// YOLODetectionHead_66675072303247
// MI455X (gfx1250) — matrix operand order repaired and checked by reading, not compiled, not run on hardware
//
#include <hip/hip_runtime.h>
#include <hip/hip_bf16.h>
#include <math.h>

// ---------------------------------------------------------------------------
// YOLO detection head for MI455X (gfx1250, wave32, WMMA + async LDS copy).
//
// Roofline: 23.4 GFLOP vs ~191 MB moved -> memory bound at 23.3 TB/s (~8 us).
// Strategy: stream each feature f32 exactly once via GLOBAL_LOAD_ASYNC_TO_LDS
// (double buffered, ASYNCcnt-tracked), cvt->f16 on fragment build, and run the
// 1x1 conv as a tiled GEMM on v_wmma_f32_16x16x32_f16 (f32 accumulation).
// ---------------------------------------------------------------------------

typedef __attribute__((ext_vector_type(16))) _Float16 v16h;
typedef __attribute__((ext_vector_type(8)))  float    v8f;

#define __AS(n) __attribute__((address_space(n)))

#if defined(__AMDGCN__) && __has_builtin(__builtin_amdgcn_global_load_async_to_lds_b32)
#define HAVE_ASYNC_LDS 1
#else
#define HAVE_ASYNC_LDS 0
#endif

#if HAVE_ASYNC_LDS
#if __has_builtin(__builtin_amdgcn_s_wait_asynccnt)
#define WAIT_ASYNC() __builtin_amdgcn_s_wait_asynccnt(0)
#else
#define WAIT_ASYNC() asm volatile("s_wait_asynccnt 0" ::: "memory")
#endif
#else
#define WAIT_ASYNC()
#endif

#define NBLK   32          // N columns per workgroup
#define KCH    32          // K chunk per iteration (WMMA K)
#define BSTR   36          // LDS feature-chunk row stride (dwords), 16B-aligned, bank-spread
#define PSTR   33          // LDS output row stride (dwords), bank-conflict padding
#define NOUT   85          // 80 classes + 5
#define NDET   25200       // total detections per batch image

__device__ inline v16h cvt16(float4 q0, float4 q1, float4 q2, float4 q3) {
    v16h v;
    v[0]  = (_Float16)q0.x; v[1]  = (_Float16)q0.y; v[2]  = (_Float16)q0.z; v[3]  = (_Float16)q0.w;
    v[4]  = (_Float16)q1.x; v[5]  = (_Float16)q1.y; v[6]  = (_Float16)q1.z; v[7]  = (_Float16)q1.w;
    v[8]  = (_Float16)q2.x; v[9]  = (_Float16)q2.y; v[10] = (_Float16)q2.z; v[11] = (_Float16)q2.w;
    v[12] = (_Float16)q3.x; v[13] = (_Float16)q3.y; v[14] = (_Float16)q3.z; v[15] = (_Float16)q3.w;
    return v;
}

__device__ inline float sigm(float x) { return 1.0f / (1.0f + expf(-x)); }

// Stage this thread's 4 dwords of a 32x32 K-chunk: LDS[.. + u*8] = G[.. + u*8*HW]
__device__ inline void stage_chunk(float* dst, const float* gsrc, size_t off8) {
#if HAVE_ASYNC_LDS
    #pragma unroll
    for (int u = 0; u < 4; ++u) {
        __builtin_amdgcn_global_load_async_to_lds_b32(
            (__AS(1) int*)(gsrc + (size_t)u * off8),
            (__AS(3) int*)(dst + u * 8),
            0, 0);
    }
#else
    float tmp[4];
    #pragma unroll
    for (int u = 0; u < 4; ++u) tmp[u] = gsrc[(size_t)u * off8];
    #pragma unroll
    for (int u = 0; u < 4; ++u) dst[u * 8] = tmp[u];
#endif
}

__global__ __launch_bounds__(256) void yolo_head_kernel(
    const float* __restrict__ feat,   // [16, C, H, W]
    const float* __restrict__ wgt,    // [255, C] row-major
    const float* __restrict__ bias,   // [255]
    float* __restrict__ out,          // [16, 25200, 5]
    int C, int H, int W, float stride_,
    float a0w, float a0h, float a1w, float a1h, float a2w, float a2h,
    int outOff)
{
    __shared__ __align__(16) float lb[2][NBLK * BSTR]; // f32 feature chunks [buf][col][k]
    __shared__ float lp[256 * PSTR];                   // f32 logits [o][col]

    const int t     = threadIdx.x;
    const int lane  = t & 31;
    const int wv    = t >> 5;        // wave id 0..7
    const int nl    = lane & 15;     // sub-lane (N index for B/C, M index for A)
    const int hi    = lane >> 4;     // lane half
    const int HWp   = H * W;
    const int nbase = blockIdx.x * NBLK;

    // --- per-thread staging role: col = t&31, k-index = (t>>5) + 8u ----------
    const int scol = t & (NBLK - 1);
    const int skq  = t >> 5;                     // 0..7
    int    n_s  = nbase + scol;
    int    b_s  = n_s / HWp;
    int    hw_s = n_s - b_s * HWp;
    size_t colbase = (size_t)b_s * (size_t)C * (size_t)HWp + (size_t)hw_s;

    const size_t off8  = (size_t)8  * (size_t)HWp;   // 8 k-rows
    const size_t offCh = (size_t)KCH * (size_t)HWp;  // one K chunk

    const float* gcur = feat + colbase + (size_t)skq * (size_t)HWp;
    float* lrow0 = &lb[0][scol * BSTR + skq];
    float* lrow1 = &lb[1][scol * BSTR + skq];

    // weight row offsets (32-bit, rederived from the kernarg base every chunk
    // so the loads stay addrspace(1) global_load_b128 in saddr+voffset form)
    int aoff0, aoff1;
    {
        int m0 = (wv * 2 + 0) * 16 + nl;
        int m1 = (wv * 2 + 1) * 16 + nl;
        int r0 = m0 > 254 ? 254 : m0;
        int r1 = m1 > 254 ? 254 : m1;
        aoff0 = r0 * C + 8 * hi;
        aoff1 = r1 * C + 8 * hi;
    }

    v8f zero8 = {0.f, 0.f, 0.f, 0.f, 0.f, 0.f, 0.f, 0.f};
    v8f acc[2][2] = { { zero8, zero8 }, { zero8, zero8 } };

    const int nchunks = C / KCH;

    // prologue: kick chunk 0 into buffer 0
    stage_chunk(lrow0, gcur, off8);
    gcur += offCh;

    for (int kc = 0; kc < nchunks; ++kc) {
        const int cur = kc & 1;

        // my chunk-kc async writes done, then WG-wide: everyone's done + the
        // buffer we are about to overwrite has been fully consumed.
        WAIT_ASYNC();
        __syncthreads();

        // overlap: launch chunk kc+1 into the other buffer while computing kc
        if (kc + 1 < nchunks) {
            stage_chunk(cur ? lrow0 : lrow1, gcur, off8);
            gcur += offCh;
        }

        const float* lbc = lb[cur];

        // ---- B fragments (32x16 f16, K x N): lane holds k = 8*hi + 0..7, 16 + 8*hi + 0..7 ----
        v16h bf[2];
        #pragma unroll
        for (int nt = 0; nt < 2; ++nt) {
            const float* bp = lbc + (nt * 16 + nl) * BSTR + hi * 8;
            float4 q0 = *(const float4*)(bp + 0);
            float4 q1 = *(const float4*)(bp + 4);
            float4 q2 = *(const float4*)(bp + 16);
            float4 q3 = *(const float4*)(bp + 20);
            bf[nt] = cvt16(q0, q1, q2, q3);
        }

        // ---- A fragments (16x32 f16, M x K) + 4 WMMAs -------------------------
        {
            float4 r0 = *(const float4*)(wgt + aoff0 + 0);   // k = 8hi..+7
            float4 r1 = *(const float4*)(wgt + aoff0 + 4);
            float4 r2 = *(const float4*)(wgt + aoff0 + 16);  // k = 16+8hi..+7
            float4 r3 = *(const float4*)(wgt + aoff0 + 20);
            v16h av = cvt16(r0, r1, r2, r3);
            acc[0][0] = __builtin_amdgcn_wmma_f32_16x16x32_f16(
                false, av, false, bf[0], (short)0, acc[0][0], false, false);
            acc[0][1] = __builtin_amdgcn_wmma_f32_16x16x32_f16(
                false, av, false, bf[1], (short)0, acc[0][1], false, false);
        }
        {
            float4 r0 = *(const float4*)(wgt + aoff1 + 0);
            float4 r1 = *(const float4*)(wgt + aoff1 + 4);
            float4 r2 = *(const float4*)(wgt + aoff1 + 16);
            float4 r3 = *(const float4*)(wgt + aoff1 + 20);
            v16h av = cvt16(r0, r1, r2, r3);
            acc[1][0] = __builtin_amdgcn_wmma_f32_16x16x32_f16(
                false, av, false, bf[0], (short)0, acc[1][0], false, false);
            acc[1][1] = __builtin_amdgcn_wmma_f32_16x16x32_f16(
                false, av, false, bf[1], (short)0, acc[1][1], false, false);
        }
        aoff0 += KCH;
        aoff1 += KCH;
    }

    // ---- spill accumulators to LDS: C/D layout VGPR r -> M = r + 8*hi --------
    __syncthreads();
    #pragma unroll
    for (int mi = 0; mi < 2; ++mi)
        #pragma unroll
        for (int nt = 0; nt < 2; ++nt)
            #pragma unroll
            for (int r = 0; r < 8; ++r) {
                int m   = (wv * 2 + mi) * 16 + r + hi * 8;
                int col = nt * 16 + nl;
                lp[m * PSTR + col] = acc[mi][nt][r];
            }
    __syncthreads();

    // ---- decode: 32 cols x 3 anchors -----------------------------------------
    if (t < 96) {
        const int col = t & 31;
        const int a   = t >> 5;
        const int n   = nbase + col;
        const int b   = n / HWp;
        const int hw  = n - b * HWp;
        const int h   = hw / W;
        const int w   = hw - h * W;
        const int ob  = a * NOUT;

        float l0 = lp[(ob + 0) * PSTR + col] + bias[ob + 0];
        float l1 = lp[(ob + 1) * PSTR + col] + bias[ob + 1];
        float l2 = lp[(ob + 2) * PSTR + col] + bias[ob + 2];
        float l3 = lp[(ob + 3) * PSTR + col] + bias[ob + 3];
        float l4 = lp[(ob + 4) * PSTR + col] + bias[ob + 4];

        float maxl = -3.4e38f;                 // sigmoid monotone: max logits first
        #pragma unroll 4
        for (int j = 5; j < NOUT; ++j) {
            float v = lp[(ob + j) * PSTR + col] + bias[ob + j];
            maxl = fmaxf(maxl, v);
        }

        float sx = sigm(l0), sy = sigm(l1), sw = sigm(l2), sh = sigm(l3);
        float score = sigm(l4) * sigm(maxl);

        float aw = (a == 0) ? a0w : ((a == 1) ? a1w : a2w);   // anchors in px:
        float ah = (a == 0) ? a0h : ((a == 1) ? a1h : a2h);   // (anchor/stride)*stride

        float cx = (sx * 2.0f - 0.5f + (float)w) * stride_;
        float cy = (sy * 2.0f - 0.5f + (float)h) * stride_;
        float bw = (sw * 2.0f) * (sw * 2.0f) * aw;
        float bh = (sh * 2.0f) * (sh * 2.0f) * ah;

        size_t oi = ((size_t)b * NDET + (size_t)outOff + (size_t)a * HWp + (size_t)hw) * 5;
        out[oi + 0] = cx - bw * 0.5f;
        out[oi + 1] = cy - bh * 0.5f;
        out[oi + 2] = cx + bw * 0.5f;
        out[oi + 3] = cy + bh * 0.5f;
        out[oi + 4] = score;
    }
}

extern "C" void kernel_launch(void* const* d_in, const int* in_sizes, int n_in,
                              void* d_out, int out_size, void* d_ws, size_t ws_size,
                              hipStream_t stream) {
    (void)in_sizes; (void)n_in; (void)out_size; (void)d_ws; (void)ws_size;
    const float* f0 = (const float*)d_in[0];
    const float* w0 = (const float*)d_in[1];
    const float* b0 = (const float*)d_in[2];
    const float* f1 = (const float*)d_in[3];
    const float* w1 = (const float*)d_in[4];
    const float* b1 = (const float*)d_in[5];
    const float* f2 = (const float*)d_in[6];
    const float* w2 = (const float*)d_in[7];
    const float* b2 = (const float*)d_in[8];
    float* out = (float*)d_out;

    dim3 blk(256);
    // scale 0: C=256, 80x80, stride 8, anchors (10,13)(16,30)(33,23), det offset 0
    yolo_head_kernel<<<dim3(16 * 80 * 80 / NBLK), blk, 0, stream>>>(
        f0, w0, b0, out, 256, 80, 80, 8.0f,
        10.f, 13.f, 16.f, 30.f, 33.f, 23.f, 0);
    // scale 1: C=512, 40x40, stride 16, anchors (30,61)(62,45)(59,119), offset 19200
    yolo_head_kernel<<<dim3(16 * 40 * 40 / NBLK), blk, 0, stream>>>(
        f1, w1, b1, out, 512, 40, 40, 16.0f,
        30.f, 61.f, 62.f, 45.f, 59.f, 119.f, 19200);
    // scale 2: C=1024, 20x20, stride 32, anchors (116,90)(156,198)(373,326), offset 24000
    yolo_head_kernel<<<dim3(16 * 20 * 20 / NBLK), blk, 0, stream>>>(
        f2, w2, b2, out, 1024, 20, 20, 32.0f,
        116.f, 90.f, 156.f, 198.f, 373.f, 326.f, 24000);
}
